// Psi_88313117540828
// MI455X (gfx1250) — matrix operand order repaired and checked by reading, not compiled, not run on hardware
//
#include <hip/hip_runtime.h>

// ---------------------------------------------------------------------------
// Heterogeneous graph transformer (2 layers, 3 relations) for MI455X / gfx1250.
// Dense 128x128 projections: V_WMMA_F32_16X16X32_F16, one wave = 16x128 output
// (one A fragment reused across 8 B column tiles -> 32 WMMAs/wave, A read 1x).
// Edge softmax/aggregation: wave32 per edge, scalar edge indices, coalesced
// 256B node-row gathers from the 192MB L2-resident feature tables.
// ---------------------------------------------------------------------------

static const int N_NODES = 100000;   // N_CURR == N_ACT
static const int E_EDGES = 600000;
static const int HID     = 128;

typedef __attribute__((ext_vector_type(16))) _Float16 v16h;
typedef __attribute__((ext_vector_type(8)))  float    v8f;
typedef __attribute__((ext_vector_type(2)))  _Float16 h2;

__device__ __forceinline__ float silu_f(float y) { return y / (1.0f + __expf(-y)); }

// ---- float atomic max via CAS (scores may be negative) ---------------------
__device__ __forceinline__ void atomicMaxF(float* addr, float val) {
  unsigned int* ua = (unsigned int*)addr;
  unsigned int old = __float_as_uint(*addr);
  while (__uint_as_float(old) < val) {
    unsigned int assumed = old;
    old = atomicCAS(ua, assumed, __float_as_uint(val));
    if (old == assumed) break;
  }
}

// ---- weight pack: Wt[n*128+k] = (f16) W[k*128+n]  (col-major f16) ----------
__global__ void pack_w_kernel(const float* __restrict__ W, _Float16* __restrict__ Wt) {
  int n = blockIdx.x, k = threadIdx.x;
  Wt[n * 128 + k] = (_Float16)W[k * 128 + n];
}

// ---- WMMA GEMM: O[N,128] = mode(A[N,128] @ W + b) --------------------------
// Block = 256 threads = 8 waves; each wave owns a 16-row tile and produces all
// 128 output columns: per K-step, one A fragment feeds 8 WMMAs (8 col tiles).
enum { GEMM_F32 = 0, GEMM_F32_SILU = 1, GEMM_F32_ACC = 2, GEMM_F16 = 3 };

template <int MODE>
__global__ __launch_bounds__(256) void gemm128_wmma(const float* __restrict__ A,
                                                    const _Float16* __restrict__ Wt,
                                                    const float* __restrict__ bias,
                                                    void* __restrict__ Out) {
  const int lane = threadIdx.x & 31;
  const int wave = threadIdx.x >> 5;
  const int row0 = (blockIdx.x * 8 + wave) * 16;
  if (row0 >= N_NODES) return;            // tail waves idle (6250 tiles / 8)
  const int rA = row0 + (lane & 15);
  const int hs = lane >> 4;               // lane half-select
  const int cB = lane & 15;

  v8f c[8];
#pragma unroll
  for (int ct = 0; ct < 8; ++ct) c[ct] = (v8f){};

#pragma unroll
  for (int kb = 0; kb < 128; kb += 32) {
    // A fragment: 16x32 f16, contiguous k order — lanes 0-15 hold K {0..15}, lanes 16-31 {16..31}
    const float* ap = A + (size_t)rA * 128 + kb;
    float4 a0 = *(const float4*)(ap + 16 * hs);
    float4 a1 = *(const float4*)(ap + 16 * hs + 4);
    float4 a2 = *(const float4*)(ap + 16 * hs + 8);
    float4 a3 = *(const float4*)(ap + 16 * hs + 12);
    v16h a;
    a[0]  = (_Float16)a0.x; a[1]  = (_Float16)a0.y; a[2]  = (_Float16)a0.z; a[3]  = (_Float16)a0.w;
    a[4]  = (_Float16)a1.x; a[5]  = (_Float16)a1.y; a[6]  = (_Float16)a1.z; a[7]  = (_Float16)a1.w;
    a[8]  = (_Float16)a2.x; a[9]  = (_Float16)a2.y; a[10] = (_Float16)a2.z; a[11] = (_Float16)a2.w;
    a[12] = (_Float16)a3.x; a[13] = (_Float16)a3.y; a[14] = (_Float16)a3.z; a[15] = (_Float16)a3.w;
    // B fragments: 32x16 f16, k order as A — lane = column, 16 consecutive K (32B contiguous, L0-hot)
    const _Float16* bp = Wt + (size_t)cB * 128 + kb + 16 * hs;
#pragma unroll
    for (int ct = 0; ct < 8; ++ct) {
      v16h b = *(const v16h*)(bp + (size_t)ct * (16 * 128));
      c[ct] = __builtin_amdgcn_wmma_f32_16x16x32_f16(false, a, false, b, (short)0, c[ct],
                                                     false, false);
    }
  }

#pragma unroll
  for (int ct = 0; ct < 8; ++ct) {
    const int   colB = ct * 16 + cB;
    const float bcol = bias[colB];
#pragma unroll
    for (int r = 0; r < 8; ++r) {
      int    row = row0 + r + 8 * hs;  // C layout: VGPR r -> M=r (lanes 0-15) / M=r+8 (lanes 16-31)
      size_t idx = (size_t)row * 128 + colB;
      float  v   = c[ct][r] + bcol;
      if (MODE == GEMM_F32_SILU)      ((float*)Out)[idx] = silu_f(v);
      else if (MODE == GEMM_F32_ACC)  ((float*)Out)[idx] += v;  // disjoint tiles: no atomics
      else if (MODE == GEMM_F16)      ((_Float16*)Out)[idx] = (_Float16)v;
      else                            ((float*)Out)[idx] = v;
    }
  }
}

// ---- segment-softmax state init --------------------------------------------
__global__ void init_md_kernel(float* __restrict__ m, float* __restrict__ denom, int n4) {
  int t = blockIdx.x * blockDim.x + threadIdx.x;
  if (t < n4) { m[t] = -3.4e38f; denom[t] = 0.0f; }
}

// ---- wave per edge: q.k logits for all 4 heads + segment max ---------------
// Lane handles channels [4*lane, 4*lane+4); head = lane/8; 8-lane shfl reduce.
__global__ __launch_bounds__(256) void edge_score_kernel(const int* __restrict__ src,
                                                         const int* __restrict__ dst,
                                                         const _Float16* __restrict__ q16,
                                                         const _Float16* __restrict__ k16,
                                                         float* __restrict__ sc,
                                                         float* __restrict__ m) {
  const int e    = blockIdx.x * 8 + (threadIdx.x >> 5);  // wave-uniform -> scalar loads
  const int lane = threadIdx.x & 31;
  const int d = dst[e], s = src[e];
  const h2* qp = (const h2*)(q16 + (size_t)d * 128 + lane * 4);
  const h2* kp = (const h2*)(k16 + (size_t)s * 128 + lane * 4);
  h2 q0 = qp[0], q1 = qp[1], k0 = kp[0], k1 = kp[1];
  float part = (float)q0.x * (float)k0.x + (float)q0.y * (float)k0.y +
               (float)q1.x * (float)k1.x + (float)q1.y * (float)k1.y;
  part += __shfl_xor(part, 1, 32);
  part += __shfl_xor(part, 2, 32);
  part += __shfl_xor(part, 4, 32);
  if ((lane & 7) == 0) {
    int   h = lane >> 3;
    float v = part * 0.17677669529663687f;  // 1/sqrt(HEAD_DIM=32)
    sc[e * 4 + h] = v;
    atomicMaxF(&m[d * 4 + h], v);
  }
}

// ---- exp(score - max) + segment sum ----------------------------------------
__global__ __launch_bounds__(256) void edge_exp_kernel(const int* __restrict__ dst,
                                                       float* __restrict__ sc,
                                                       const float* __restrict__ m,
                                                       float* __restrict__ denom, int e4) {
  int t = blockIdx.x * blockDim.x + threadIdx.x;
  if (t >= e4) return;
  int e = t >> 2, h = t & 3;
  int d = dst[e];
  float v = __expf(sc[t] - m[d * 4 + h]);
  sc[t] = v;
  atomicAdd(&denom[d * 4 + h], v);
}

// ---- wave per edge: agg[dst] += alpha_h * v[src] ---------------------------
__global__ __launch_bounds__(256) void edge_agg_kernel(const int* __restrict__ src,
                                                       const int* __restrict__ dst,
                                                       const float* __restrict__ sc,
                                                       const float* __restrict__ denom,
                                                       const _Float16* __restrict__ v16,
                                                       float* __restrict__ agg) {
  const int e    = blockIdx.x * 8 + (threadIdx.x >> 5);  // wave-uniform -> scalar loads
  const int lane = threadIdx.x & 31;
  const int d = dst[e], s = src[e];
  const int c0 = lane * 4;
  const int h  = lane >> 3;
  float alpha = sc[e * 4 + h] / fmaxf(denom[d * 4 + h], 1e-16f);
  const h2* vp = (const h2*)(v16 + (size_t)s * 128 + c0);
  h2 v0 = vp[0], v1 = vp[1];
  float* ap = agg + (size_t)d * 128 + c0;
  atomicAdd(ap + 0, alpha * (float)v0.x);
  atomicAdd(ap + 1, alpha * (float)v0.y);
  atomicAdd(ap + 2, alpha * (float)v1.x);
  atomicAdd(ap + 3, alpha * (float)v1.y);
}

// ---- LayerNorm + SiLU, one wave32 per row ----------------------------------
__global__ __launch_bounds__(256) void ln_silu_kernel(const float* __restrict__ X,
                                                      const float* __restrict__ g,
                                                      const float* __restrict__ b,
                                                      float* __restrict__ Out) {
  int row  = blockIdx.x * 8 + (threadIdx.x >> 5);
  int lane = threadIdx.x & 31;
  const float4 xv = *(const float4*)(X + (size_t)row * 128 + lane * 4);
  float s = xv.x + xv.y + xv.z + xv.w;
  float q = xv.x * xv.x + xv.y * xv.y + xv.z * xv.z + xv.w * xv.w;
#pragma unroll
  for (int off = 16; off > 0; off >>= 1) {
    s += __shfl_xor(s, off, 32);
    q += __shfl_xor(q, off, 32);
  }
  float mu   = s * (1.0f / 128.0f);
  float var  = q * (1.0f / 128.0f) - mu * mu;
  float rstd = rsqrtf(var + 1e-5f);
  float4 gv = *(const float4*)(g + lane * 4);
  float4 bv = *(const float4*)(b + lane * 4);
  float4 o;
  o.x = silu_f(gv.x * (xv.x - mu) * rstd + bv.x);
  o.y = silu_f(gv.y * (xv.y - mu) * rstd + bv.y);
  o.z = silu_f(gv.z * (xv.z - mu) * rstd + bv.z);
  o.w = silu_f(gv.w * (xv.w - mu) * rstd + bv.w);
  *(float4*)(Out + (size_t)row * 128 + lane * 4) = o;
}

// ===========================================================================
// Host orchestration
// ===========================================================================
struct LinP { const float* W; const float* b; };
struct Resolved {
  const float *x_curr, *x_act;
  LinP in_curr, in_act, out_act;
  const float *ln_curr_g, *ln_curr_b, *ln_act_g, *ln_act_b;
  LinP rel[2][3][4];  // [layer][0=c2a,1=tmp,2=a2c][0=q,1=k,2=v,3=s]
  const int *ei_c2a, *ei_a2c, *ei_tmp;
};

static void launch_gemm(int mode, const float* A, const _Float16* Wt, const float* bias,
                        void* out, hipStream_t s) {
  dim3 g((N_NODES / 16 + 7) / 8), b(256);
  switch (mode) {
    case GEMM_F32:      gemm128_wmma<GEMM_F32><<<g, b, 0, s>>>(A, Wt, bias, out); break;
    case GEMM_F32_SILU: gemm128_wmma<GEMM_F32_SILU><<<g, b, 0, s>>>(A, Wt, bias, out); break;
    case GEMM_F32_ACC:  gemm128_wmma<GEMM_F32_ACC><<<g, b, 0, s>>>(A, Wt, bias, out); break;
    default:            gemm128_wmma<GEMM_F16><<<g, b, 0, s>>>(A, Wt, bias, out); break;
  }
}

extern "C" void kernel_launch(void* const* d_in, const int* in_sizes, int n_in,
                              void* d_out, int out_size, void* d_ws, size_t ws_size,
                              hipStream_t stream) {
  (void)n_in; (void)out_size; (void)ws_size;
  auto F = [&](int i) { return (const float*)d_in[i]; };
  auto I = [&](int i) { return (const int*)d_in[i]; };

  Resolved R;
  if (in_sizes[0] == N_NODES * 128) {
    // Top-level + nested insertion order:
    // x_curr,x_act, in_curr{W,b}, in_act{W,b},
    // layers[l]{ c2a{q,k,v,s}{W,b}, tmp{...}, a2c{...} },
    // ln_curr{g,b}, ln_act{g,b}, out_act{W,b}, ei_c2a, ei_a2c, ei_tmp
    R.x_curr = F(0); R.x_act = F(1);
    R.in_curr = {F(2), F(3)}; R.in_act = {F(4), F(5)};
    for (int l = 0; l < 2; ++l) {
      int base = 6 + l * 24;
      for (int r = 0; r < 3; ++r)          // rel order c2a,tmp,a2c matches our relid
        for (int p = 0; p < 4; ++p)        // q,k,v,s
          R.rel[l][r][p] = {F(base + r * 8 + p * 2), F(base + r * 8 + p * 2 + 1)};
    }
    R.ln_curr_g = F(54); R.ln_curr_b = F(55);
    R.ln_act_g  = F(56); R.ln_act_b  = F(57);
    R.out_act = {F(58), F(59)};
    R.ei_c2a = I(60); R.ei_a2c = I(61); R.ei_tmp = I(62);
  } else {
    // jax pytree (sorted-key) order:
    // ei_a2c, ei_c2a, ei_tmp, in_act{W,b}, in_curr{W,b},
    // layers[l]{ a2c{k,q,s,v}{W,b}, c2a{...}, tmp{...} },
    // ln_act{b,g}, ln_curr{b,g}, out_act{W,b}, x_act, x_curr
    R.ei_a2c = I(0); R.ei_c2a = I(1); R.ei_tmp = I(2);
    R.in_act = {F(3), F(4)}; R.in_curr = {F(5), F(6)};
    const int relmap[3] = {2, 0, 1};  // sorted pos {a2c,c2a,tmp} -> our relid {c2a,tmp,a2c}
    for (int l = 0; l < 2; ++l) {
      int base = 7 + l * 24;
      for (int rs = 0; rs < 3; ++rs) {
        int r  = relmap[rs];
        int rb = base + rs * 8;
        R.rel[l][r][0] = {F(rb + 2), F(rb + 3)};  // q
        R.rel[l][r][1] = {F(rb + 0), F(rb + 1)};  // k
        R.rel[l][r][2] = {F(rb + 6), F(rb + 7)};  // v
        R.rel[l][r][3] = {F(rb + 4), F(rb + 5)};  // s
      }
    }
    R.ln_act_b  = F(55); R.ln_act_g  = F(56);
    R.ln_curr_b = F(57); R.ln_curr_g = F(58);
    R.out_act = {F(59), F(60)};
    R.x_act = F(61); R.x_curr = F(62);
  }

  // ---- workspace carve-up --------------------------------------------------
  const size_t NF = (size_t)N_NODES * HID;
  char* wp = (char*)d_ws;
  auto carve = [&](size_t bytes) { void* r = wp; wp += (bytes + 255) & ~(size_t)255; return r; };
  float*    xc    = (float*)carve(NF * 4);
  float*    xa    = (float*)carve(NF * 4);
  float*    agg_a = (float*)carve(NF * 4);
  float*    agg_c = (float*)carve(NF * 4);
  _Float16* q16   = (_Float16*)carve(NF * 2);
  _Float16* k16   = (_Float16*)carve(NF * 2);
  _Float16* v16   = (_Float16*)carve(NF * 2);
  float*    sc    = (float*)carve((size_t)E_EDGES * 4 * 4);
  float*    mbuf  = (float*)carve((size_t)N_NODES * 4 * 4);
  float*    dbuf  = (float*)carve((size_t)N_NODES * 4 * 4);
  _Float16* w16   = (_Float16*)carve((size_t)27 * 16384 * 2);

  // ---- pack all 27 weight matrices to f16 col-major ------------------------
  auto wslot = [&](int slot) { return w16 + (size_t)slot * 16384; };
  pack_w_kernel<<<128, 128, 0, stream>>>(R.in_curr.W, wslot(0));
  pack_w_kernel<<<128, 128, 0, stream>>>(R.in_act.W,  wslot(1));
  pack_w_kernel<<<128, 128, 0, stream>>>(R.out_act.W, wslot(2));
  for (int l = 0; l < 2; ++l)
    for (int r = 0; r < 3; ++r)
      for (int p = 0; p < 4; ++p)
        pack_w_kernel<<<128, 128, 0, stream>>>(R.rel[l][r][p].W, wslot(3 + l * 12 + r * 4 + p));

  const int n4 = N_NODES * 4, e4 = E_EDGES * 4;
  const int gInit = (n4 + 255) / 256;
  const int gE4   = (e4 + 255) / 256;
  const int gEdge = E_EDGES / 8;  // wave-per-edge kernels, 8 waves/block

  auto tconv = [&](const float* xsrc, const float* xdst, const int* ei, int l, int r,
                   float* agg, bool acc_s) {
    int s3 = 3 + l * 12 + r * 4;
    launch_gemm(GEMM_F16, xdst, wslot(s3 + 0), R.rel[l][r][0].b, q16, stream);  // q(dst)
    launch_gemm(GEMM_F16, xsrc, wslot(s3 + 1), R.rel[l][r][1].b, k16, stream);  // k(src)
    launch_gemm(GEMM_F16, xsrc, wslot(s3 + 2), R.rel[l][r][2].b, v16, stream);  // v(src)
    launch_gemm(acc_s ? GEMM_F32_ACC : GEMM_F32, xdst, wslot(s3 + 3),
                R.rel[l][r][3].b, agg, stream);                                 // agg (+)= s(dst)
    init_md_kernel<<<gInit, 256, 0, stream>>>(mbuf, dbuf, n4);
    const int* src = ei;
    const int* dst = ei + E_EDGES;
    edge_score_kernel<<<gEdge, 256, 0, stream>>>(src, dst, q16, k16, sc, mbuf);
    edge_exp_kernel<<<gE4, 256, 0, stream>>>(dst, sc, mbuf, dbuf, e4);
    edge_agg_kernel<<<gEdge, 256, 0, stream>>>(src, dst, sc, dbuf, v16, agg);
  };

  // ---- pipeline ------------------------------------------------------------
  launch_gemm(GEMM_F32_SILU, R.x_curr, wslot(0), R.in_curr.b, xc, stream);
  launch_gemm(GEMM_F32_SILU, R.x_act,  wslot(1), R.in_act.b,  xa, stream);

  for (int l = 0; l < 2; ++l) {
    tconv(xc, xa, R.ei_c2a, l, 0, agg_a, false);  // curr -> act
    tconv(xa, xa, R.ei_tmp, l, 1, agg_a, true);   // act -> act (summed)
    tconv(xa, xc, R.ei_a2c, l, 2, agg_c, false);  // act -> curr
    ln_silu_kernel<<<N_NODES / 8, 256, 0, stream>>>(agg_c, R.ln_curr_g, R.ln_curr_b, xc);
    ln_silu_kernel<<<N_NODES / 8, 256, 0, stream>>>(agg_a, R.ln_act_g,  R.ln_act_b,  xa);
  }

  launch_gemm(GEMM_F32, xa, wslot(2), R.out_act.b, (float*)d_out, stream);
}
